// SpatialCorrelativeLoss_66125316489536
// MI455X (gfx1250) — matrix operand order repaired and checked by reading, not compiled, not run on hardware
//
#include <hip/hip_runtime.h>
#include <hip/hip_bf16.h>

typedef __attribute__((ext_vector_type(16))) _Float16 v16h;
typedef __attribute__((ext_vector_type(8)))  _Float16 v8h;   // naturally 16B aligned
typedef __attribute__((ext_vector_type(8)))  float    v8f;

#define K_C    512
#define K_C4   128
#define K_W    128
#define K_H    128
#define K_WH   16384
#define K_B    4
#define K_NUM  64

// workspace layout (bytes)
#define OFF_FEAT  0ull                               // 2*4*128*16384 f32 = 67108864
#define OFF_W1F   67108864ull                        // 65536 f16
#define OFF_W2F   (OFF_W1F + 131072ull)              // 16384 f16
#define OFF_MEAN  (OFF_W2F + 32768ull)               // 1024 f32
#define OFF_SIM   (OFF_MEAN + 4096ull)               // 2*4*64*81 f32 = 165888
#define OFF_PART  (OFF_SIM + 165888ull)              // 1024*256 f32 = 1048576

#define XSLAB 2560                                   // 64 px * 40 halfs per slab buffer

// contiguous 16-bit A-matrix 16x32 layout: element j of v16h -> K offset (16*half + j) within 32-wide K tile
__device__ __forceinline__ int kof(int j, int lane) {
    int base = j;
    return base + ((lane & 16) ? 16 : 0);
}

__device__ __forceinline__ v16h load16h(const _Float16* p) {
    union { v16h v; v8h h[2]; } u;
    u.h[0] = *(const v8h*)p;        // ds/global b128
    u.h[1] = *(const v8h*)(p + 8);  // ds/global b128
    return u.v;
}

__device__ __forceinline__ unsigned pack2h(float a, float b) {
    union { _Float16 h[2]; unsigned u; } p;
    p.h[0] = (_Float16)a;
    p.h[1] = (_Float16)b;
    return p.u;
}

// ---------------- kernel 0: pack weights into WMMA A-fragment order -------------
__global__ __launch_bounds__(256) void pack_w(const float* __restrict__ W1,
                                              const float* __restrict__ W2,
                                              _Float16* __restrict__ w1f,
                                              _Float16* __restrict__ w2f) {
    int e = blockIdx.x * 256 + threadIdx.x;
    if (e < 65536) {                       // W1: 8 mtiles x 16 ktiles, frag = 512 f16
        int f = e >> 9, rem = e & 511, lane = rem >> 4, j = rem & 15;
        int mt = f >> 4, kt = f & 15;
        int M = mt * 16 + (lane & 15);
        int K = kt * 32 + kof(j, lane);
        w1f[e] = (_Float16)W1[M * K_C + K];
    } else {
        int e2 = e - 65536;                // W2: 8 mtiles x 4 ktiles
        if (e2 < 16384) {
            int f = e2 >> 9, rem = e2 & 511, lane = rem >> 4, j = rem & 15;
            int mt = f >> 2, kt = f & 3;
            int M = mt * 16 + (lane & 15);
            int K = kt * 32 + kof(j, lane);
            w2f[e2] = (_Float16)W2[M * K_C4 + K];
        }
    }
}

// ---------------- kernel 1: fused 2-layer MLP via v_wmma_f32_16x16x32_f16 ------
// block = 64 pixels x 128 out channels; 8 waves, wave w owns M-tile w, N-tiles 0..3
__global__ __launch_bounds__(256) void mlp_wmma(const float* __restrict__ fsrc,
                                                const float* __restrict__ ftgt,
                                                const _Float16* __restrict__ w1f,
                                                const _Float16* __restrict__ w2f,
                                                const float* __restrict__ b1,
                                                const float* __restrict__ b2,
                                                float* __restrict__ feat,
                                                float* __restrict__ part) {
    __shared__ _Float16 xs[2 * XSLAB];  // double-buffered [px][c0..31 + 8 pad]
    __shared__ _Float16 hs[64 * 136];   // [px][hc0..127 + 8 pad] (272B px stride)

    int blk  = blockIdx.x;              // 0..2047
    int inp  = blk >> 10;
    int tile = blk & 1023;
    int b    = tile >> 8;
    int stile = tile & 255;
    int s0   = stile * 64;
    const float* xin = (inp ? ftgt : fsrc) + (size_t)b * K_C * K_WH + s0;
    float* fout = feat + ((size_t)(inp * K_B + b) * K_C4) * K_WH + s0;

    int tid = threadIdx.x;
    int w = tid >> 5, lane = tid & 31;
    int lhalf = lane >> 4;              // 0 / 1
    int l15   = lane & 15;
    int ocbase = w * 16 + lhalf * 8;    // 8 contiguous out-channels per lane-half

    // staging: thread -> (pair of adjacent channels, 4 consecutive pixels)
    int c0  = (tid >> 4) * 2;           // 0,2,...,30
    int px0 = (tid & 15) * 4;

    // ---- layer 1: K = 512 in 16 steps of 32, double-buffered + pipelined ----
    v8f acc[4] = {};
    {
        const float* g = xin + (size_t)c0 * K_WH + px0;
        float4 ua = *(const float4*)g;
        float4 ub = *(const float4*)(g + K_WH);
        *(unsigned*)&xs[(px0 + 0) * 40 + c0] = pack2h(ua.x, ub.x);
        *(unsigned*)&xs[(px0 + 1) * 40 + c0] = pack2h(ua.y, ub.y);
        *(unsigned*)&xs[(px0 + 2) * 40 + c0] = pack2h(ua.z, ub.z);
        *(unsigned*)&xs[(px0 + 3) * 40 + c0] = pack2h(ua.w, ub.w);
    }
    __syncthreads();
#pragma unroll
    for (int kt = 0; kt < 16; ++kt) {
        const _Float16* cur = &xs[(kt & 1) * XSLAB];
        float4 na, nb;
        if (kt < 15) {                  // kick off next slab loads early
            const float* g = xin + (size_t)((kt + 1) * 32 + c0) * K_WH + px0;
            na = *(const float4*)g;
            nb = *(const float4*)(g + K_WH);
        }
        v16h a   = load16h(&w1f[(size_t)(w * 16 + kt) * 512 + lane * 16]);
        v16h bf0 = load16h(&cur[( 0 + l15) * 40 + lhalf * 16]);
        v16h bf1 = load16h(&cur[(16 + l15) * 40 + lhalf * 16]);
        v16h bf2 = load16h(&cur[(32 + l15) * 40 + lhalf * 16]);
        v16h bf3 = load16h(&cur[(48 + l15) * 40 + lhalf * 16]);
        acc[0] = __builtin_amdgcn_wmma_f32_16x16x32_f16(false, a, false, bf0, (short)0, acc[0], false, false);
        acc[1] = __builtin_amdgcn_wmma_f32_16x16x32_f16(false, a, false, bf1, (short)0, acc[1], false, false);
        acc[2] = __builtin_amdgcn_wmma_f32_16x16x32_f16(false, a, false, bf2, (short)0, acc[2], false, false);
        acc[3] = __builtin_amdgcn_wmma_f32_16x16x32_f16(false, a, false, bf3, (short)0, acc[3], false, false);
        if (kt < 15) {                  // write other buffer; current reads untouched
            _Float16* nxt = &xs[((kt + 1) & 1) * XSLAB];
            *(unsigned*)&nxt[(px0 + 0) * 40 + c0] = pack2h(na.x, nb.x);
            *(unsigned*)&nxt[(px0 + 1) * 40 + c0] = pack2h(na.y, nb.y);
            *(unsigned*)&nxt[(px0 + 2) * 40 + c0] = pack2h(na.z, nb.z);
            *(unsigned*)&nxt[(px0 + 3) * 40 + c0] = pack2h(na.w, nb.w);
        }
        __syncthreads();
    }

    // bias + relu + f16 pack into hs
    float bb1[8];
#pragma unroll
    for (int i = 0; i < 8; ++i) bb1[i] = b1[ocbase + i];
#pragma unroll
    for (int nt = 0; nt < 4; ++nt) {
        int px = nt * 16 + l15;
        v8h h;
#pragma unroll
        for (int i = 0; i < 8; ++i) h[i] = (_Float16)fmaxf(acc[nt][i] + bb1[i], 0.0f);
        *(v8h*)&hs[px * 136 + ocbase] = h;      // ds_store_b128
    }
    __syncthreads();

    // ---- layer 2: K = 128 in 4 steps of 32 ----
    v8f acc2[4] = {};
#pragma unroll
    for (int kt = 0; kt < 4; ++kt) {
        v16h a2  = load16h(&w2f[(size_t)(w * 4 + kt) * 512 + lane * 16]);
        v16h bf0 = load16h(&hs[( 0 + l15) * 136 + kt * 32 + lhalf * 16]);
        v16h bf1 = load16h(&hs[(16 + l15) * 136 + kt * 32 + lhalf * 16]);
        v16h bf2 = load16h(&hs[(32 + l15) * 136 + kt * 32 + lhalf * 16]);
        v16h bf3 = load16h(&hs[(48 + l15) * 136 + kt * 32 + lhalf * 16]);
        acc2[0] = __builtin_amdgcn_wmma_f32_16x16x32_f16(false, a2, false, bf0, (short)0, acc2[0], false, false);
        acc2[1] = __builtin_amdgcn_wmma_f32_16x16x32_f16(false, a2, false, bf1, (short)0, acc2[1], false, false);
        acc2[2] = __builtin_amdgcn_wmma_f32_16x16x32_f16(false, a2, false, bf2, (short)0, acc2[2], false, false);
        acc2[3] = __builtin_amdgcn_wmma_f32_16x16x32_f16(false, a2, false, bf3, (short)0, acc2[3], false, false);
    }

    // bias add, feat store, and per-channel partial sums (deterministic, no atomics)
    float bb2[8], psum[8];
#pragma unroll
    for (int i = 0; i < 8; ++i) { bb2[i] = b2[ocbase + i]; psum[i] = 0.0f; }
#pragma unroll
    for (int nt = 0; nt < 4; ++nt) {
        int px = nt * 16 + l15;
#pragma unroll
        for (int i = 0; i < 8; ++i) {
            float v = acc2[nt][i] + bb2[i];
            fout[(size_t)(ocbase + i) * K_WH + px] = v;
            psum[i] += v;
        }
    }
    // reduce over the 16 lanes of each half-wave (C/D layout: lanes = pixels)
#pragma unroll
    for (int i = 0; i < 8; ++i) {
        float s = psum[i];
        s += __shfl_xor(s, 1, 32);
        s += __shfl_xor(s, 2, 32);
        s += __shfl_xor(s, 4, 32);
        s += __shfl_xor(s, 8, 32);
        psum[i] = s;
    }
    if (l15 == 0) {
        size_t rowbase = ((size_t)(inp * K_B + b) * K_C4 + ocbase) * 256 + stile;
#pragma unroll
        for (int i = 0; i < 8; ++i) part[rowbase + (size_t)i * 256] = psum[i];
    }
}

// ---------------- kernel 2: finish spatial mean from per-tile partials ----------
__global__ __launch_bounds__(256) void chan_mean(const float* __restrict__ part,
                                                 float* __restrict__ mean) {
    int row = blockIdx.x;  // 0..1023 = (inp,b,oc)
    float s = part[(size_t)row * 256 + threadIdx.x];
    __shared__ float red[8];
    for (int m = 16; m; m >>= 1) s += __shfl_xor(s, m, 32);
    if ((threadIdx.x & 31) == 0) red[threadIdx.x >> 5] = s;
    __syncthreads();
    if (threadIdx.x == 0) {
        float t = 0.f;
        for (int i = 0; i < 8; ++i) t += red[i];
        mean[row] = t * (1.0f / K_WH);
    }
}

// ---------------- kernel 3: query-vs-window cosine sims ------------------------
__global__ __launch_bounds__(128) void patch_sim(const float* __restrict__ feat,
                                                 const float* __restrict__ mean,
                                                 const int* __restrict__ pids,
                                                 float* __restrict__ sims) {
    int blk = blockIdx.x;          // inp*256 + b*64 + n
    int inp = blk >> 8;
    int b   = (blk >> 6) & 3;
    int n   = blk & 63;
    int id  = pids[n];
    int posx = id / K_W, posy = id % K_W;
    int sx = min(max(posx - 4, 0), K_W - 9);
    int sy = min(max(posy - 4, 0), K_H - 9);
    const float* fb = feat + (size_t)(inp * K_B + b) * K_C4 * K_WH;
    const float* mb = mean + (inp * K_B + b) * K_C4;

    __shared__ float qh[128], mm[128], red[4];
    int t = threadIdx.x;           // channel in phase 1
    float m = mb[t];
    mm[t] = m;
    float q = fb[(size_t)t * K_WH + id] - m;
    float ss = q * q;
    for (int msk = 16; msk; msk >>= 1) ss += __shfl_xor(ss, msk, 32);
    if ((t & 31) == 0) red[t >> 5] = ss;
    __syncthreads();
    float nrm = sqrtf(red[0] + red[1] + red[2] + red[3]);
    qh[t] = q / fmaxf(nrm, 1e-12f);
    __syncthreads();
    if (t < 81) {                  // t = window position (coalesced over s)
        int s = (sx + t / 9) * K_H + (sy + t % 9);
        float vv = 0.f, qv = 0.f;
        for (int c = 0; c < 128; ++c) {
            float v = fb[(size_t)c * K_WH + s] - mm[c];
            vv += v * v;
            qv += qh[c] * v;
        }
        sims[(size_t)blk * 81 + t] = qv / fmaxf(sqrtf(vv), 1e-12f);
    }
}

// ---------------- kernel 4: top-21 threshold + masked cosine -> scalar ---------
__global__ __launch_bounds__(256) void compare_cos(const float* __restrict__ sims,
                                                   float* __restrict__ out) {
    __shared__ float tg[8][88];
    __shared__ float part[8];
    int t = threadIdx.x, w = t >> 5, lane = t & 31;
    const float* S = sims;
    const float* T = sims + 4 * 64 * 81;
    float tot = 0.f;
    for (int j = w; j < 256; j += 8) {       // all waves: 32 iterations
        int base = j * 81;
        float sv[3], tv[3];
        int e[3];
#pragma unroll
        for (int k = 0; k < 3; ++k) {
            e[k] = lane + 32 * k;
            bool ok = e[k] < 81;
            sv[k] = ok ? S[base + e[k]] : 0.f;
            tv[k] = ok ? T[base + e[k]] : 0.f;
            if (ok) tg[w][e[k]] = tv[k];
        }
        __syncthreads();
        // thr = sorted_desc[20] = max{v : count(u >= v) >= 21}
        float cand = -3.4e38f;
#pragma unroll
        for (int k = 0; k < 3; ++k)
            if (e[k] < 81) {
                int cnt = 0;
                for (int u = 0; u < 81; ++u) cnt += (tg[w][u] >= tv[k]) ? 1 : 0;
                if (cnt >= 21 && tv[k] > cand) cand = tv[k];
            }
        for (int msk = 16; msk; msk >>= 1) cand = fmaxf(cand, __shfl_xor(cand, msk, 32));
        float thr = cand;
        float dot = 0.f, n1 = 0.f, n2 = 0.f;
#pragma unroll
        for (int k = 0; k < 3; ++k)
            if (e[k] < 81) {
                bool lt = tv[k] < thr;
                float s_ = lt ? 0.f : sv[k];
                float t_ = lt ? 0.f : tv[k];
                dot += s_ * t_; n1 += s_ * s_; n2 += t_ * t_;
            }
        for (int msk = 16; msk; msk >>= 1) {
            dot += __shfl_xor(dot, msk, 32);
            n1  += __shfl_xor(n1, msk, 32);
            n2  += __shfl_xor(n2, msk, 32);
        }
        float simpos = dot / (fmaxf(sqrtf(n1), 1e-8f) * fmaxf(sqrtf(n2), 1e-8f));
        tot += fabsf(1.f - simpos);
        __syncthreads();                     // before tg overwrite next iter
    }
    if (lane == 0) part[w] = tot;
    __syncthreads();
    if (t == 0) {
        float z = 0.f;
        for (int i = 0; i < 8; ++i) z += part[i];
        out[0] = z * (1.0f / 256.0f);
    }
}

extern "C" void kernel_launch(void* const* d_in, const int* in_sizes, int n_in,
                              void* d_out, int out_size, void* d_ws, size_t ws_size,
                              hipStream_t stream) {
    const float* f_src = (const float*)d_in[0];
    const float* f_tgt = (const float*)d_in[1];
    const int*   pids  = (const int*)d_in[2];
    const float* W1    = (const float*)d_in[3];
    const float* b1    = (const float*)d_in[4];
    const float* W2    = (const float*)d_in[5];
    const float* b2    = (const float*)d_in[6];

    char* ws = (char*)d_ws;
    float*    feat = (float*)(ws + OFF_FEAT);
    _Float16* w1f  = (_Float16*)(ws + OFF_W1F);
    _Float16* w2f  = (_Float16*)(ws + OFF_W2F);
    float*    mean = (float*)(ws + OFF_MEAN);
    float*    sims = (float*)(ws + OFF_SIM);
    float*    parts = (float*)(ws + OFF_PART);

    hipLaunchKernelGGL(pack_w,      dim3(320),  dim3(256), 0, stream, W1, W2, w1f, w2f);
    hipLaunchKernelGGL(mlp_wmma,    dim3(2048), dim3(256), 0, stream,
                       f_src, f_tgt, w1f, w2f, b1, b2, feat, parts);
    hipLaunchKernelGGL(chan_mean,   dim3(1024), dim3(256), 0, stream, parts, mean);
    hipLaunchKernelGGL(patch_sim,   dim3(512),  dim3(128), 0, stream, feat, mean, pids, sims);
    hipLaunchKernelGGL(compare_cos, dim3(1),    dim3(256), 0, stream, sims, (float*)d_out);
}
